// EfficientMultiHeadAttention_46402826666165
// MI455X (gfx1250) — matrix operand order repaired and checked by reading, not compiled, not run on hardware
//
#include <hip/hip_runtime.h>
#include <hip/hip_bf16.h>

typedef __bf16 bf16;
typedef __attribute__((ext_vector_type(16))) bf16  v16bf;
typedef __attribute__((ext_vector_type(8)))  bf16  v8bf;
typedef __attribute__((ext_vector_type(4)))  bf16  v4bf;
typedef __attribute__((ext_vector_type(8)))  float v8f;
typedef __attribute__((ext_vector_type(4)))  float v4f;

#define DEV static __device__ __forceinline__

constexpr int D_MODEL = 1024;
constexpr int N_HEAD  = 16;
constexpr int HEAD    = 64;
constexpr int TSEQ    = 2048;
constexpr int BATCH   = 2;

DEV v8f zero8() {
  v8f z = {0.f, 0.f, 0.f, 0.f, 0.f, 0.f, 0.f, 0.f};
  return z;
}

// --- CDNA5 async global->LDS copy (ASYNCcnt-tracked, 16B per lane) ---------
DEV void async_copy16(unsigned int lds_byte_addr, unsigned long long gaddr) {
  asm volatile("global_load_async_to_lds_b128 %0, %1, off"
               :
               : "v"(lds_byte_addr), "v"(gaddr)
               : "memory");
}
DEV void wait_async() { asm volatile("s_wait_asynccnt 0" ::: "memory"); }

DEV unsigned int lds_addr_of(const void* p) {
  // generic pointer to LDS: low 32 bits are the DS-space byte address
  return (unsigned int)(unsigned long long)p;
}

// A-operand fragment (16x32 bf16), CDNA5 ISA layout:
// lanes 0-15 : M = lane,    halves 0..7 -> K=0..7,  halves 8..15 -> K=16..23
// lanes 16-31: M = lane-16, halves 0..7 -> K=8..15, halves 8..15 -> K=24..31
DEV v16bf load_frag_a(const bf16* row_base, int stride, int lane) {
  const int r    = lane & 15;
  const int koff = (lane >> 4) << 3;   // 0 or 8
  const bf16* p  = row_base + r * stride + koff;
  v8bf lo = *(const v8bf*)(p);
  v8bf hi = *(const v8bf*)(p + 16);
  v16bf out;
#pragma unroll
  for (int i = 0; i < 8; ++i) { out[i] = lo[i]; out[i + 8] = hi[i]; }
  return out;
}

// B-operand fragment (32x16 bf16): col_base points at the [N][K] tile;
// lanes 0-15: N = lane, K=0..7,16..23; lanes 16-31: K=8..15,24..31.
DEV v16bf load_frag_b(const bf16* col_base, int stride, int lane) {
  const int n    = lane & 15;
  const int koff = (lane >> 4) << 3;   // 0 or 8
  const bf16* p  = col_base + n * stride + koff;
  v8bf lo = *(const v8bf*)(p);
  v8bf hi = *(const v8bf*)(p + 16);
  v16bf out;
#pragma unroll
  for (int i = 0; i < 8; ++i) { out[i] = lo[i]; out[i + 8] = hi[i]; }
  return out;
}

DEV v8f wmma_bf16(v16bf a, v16bf b, v8f c) {
  return __builtin_amdgcn_wmma_f32_16x16x32_bf16(false, a, false, b, (short)0, c,
                                                 false, false);
}

DEV v4bf cvt4(v4f f) {
  v4bf b;
#pragma unroll
  for (int j = 0; j < 4; ++j) b[j] = (bf16)f[j];
  return b;
}

// ---------------------------------------------------------------------------
// GEMM: out(M,N) = A(M,K) * W(N,K)^T
// MODE 0: A=f32; scatter into q/k/v bf16 (B,H,T,HEAD), q scaled by 1/8
// MODE 1: A=bf16 (async-staged); f32 output + bias
// ---------------------------------------------------------------------------
template <typename AT, int MODE>
__global__ __launch_bounds__(256) void gemm_wmma_kernel(
    const AT* __restrict__ A, const float* __restrict__ W,
    int M, int N, int K,
    bf16* __restrict__ q_ws, bf16* __restrict__ k_ws, bf16* __restrict__ v_ws,
    const float* __restrict__ bias, float* __restrict__ out) {
  constexpr int BM = 128, BN = 128, BK = 64;
  constexpr int LDA = BK + 8;  // +16B pad: bank spread, keeps 16B alignment
  __shared__ bf16 As[BM][LDA];
  __shared__ bf16 Bs[BN][LDA];

  const int tid  = threadIdx.x;
  const int lane = tid & 31;
  const int wv   = tid >> 5;      // 8 waves
  const int wm   = wv & 3;        // 4 row groups of 32
  const int wn   = wv >> 2;       // 2 col groups of 64
  const int bm   = blockIdx.x * BM;
  const int bn   = blockIdx.y * BN;
  const unsigned int as_base = lds_addr_of(&As[0][0]);

  v8f acc[2][4];
#pragma unroll
  for (int i = 0; i < 2; ++i)
#pragma unroll
    for (int j = 0; j < 4; ++j) acc[i][j] = zero8();

  for (int k0 = 0; k0 < K; k0 += BK) {
    if constexpr (MODE == 1) {
      // A already bf16: async DMA 16B chunks straight into LDS
#pragma unroll
      for (int i = 0; i < 4; ++i) {
        int idx = tid + i * 256;
        int r = idx >> 3, ch = idx & 7;  // 8 x 16B chunks per 64-elem row
        async_copy16(as_base + (unsigned)(r * LDA * 2 + ch * 16),
                     (unsigned long long)(const char*)A +
                         ((size_t)(bm + r) * K + k0 + ch * 8) * sizeof(bf16));
      }
    } else {
      // f32 -> bf16, vectorized float4 loads / b64 LDS stores
#pragma unroll
      for (int i = 0; i < 8; ++i) {
        int idx = tid + i * 256;
        int r = idx >> 4, c4 = (idx & 15) << 2;
        v4f a4 = *(const v4f*)((const float*)A + (size_t)(bm + r) * K + k0 + c4);
        *(v4bf*)&As[r][c4] = cvt4(a4);
      }
    }
    // W staging (always f32)
#pragma unroll
    for (int i = 0; i < 8; ++i) {
      int idx = tid + i * 256;
      int r = idx >> 4, c4 = (idx & 15) << 2;
      v4f w4 = *(const v4f*)(W + (size_t)(bn + r) * K + k0 + c4);
      *(v4bf*)&Bs[r][c4] = cvt4(w4);
    }
    if (k0 + BK < K) {  // hint next tiles toward L2/WGP$ (global_prefetch_b8)
      __builtin_prefetch((const char*)A +
                             ((size_t)(bm + (tid >> 1)) * K + k0 + BK +
                              (tid & 1) * 32) * sizeof(AT), 0, 1);
      __builtin_prefetch(W + (size_t)(bn + (tid >> 1)) * K + k0 + BK +
                             (tid & 1) * 32, 0, 1);
    }
    if constexpr (MODE == 1) wait_async();
    __syncthreads();

#pragma unroll
    for (int kk = 0; kk < BK; kk += 32) {
      v16bf a0 = load_frag_a(&As[wm * 32 + 0][kk], LDA, lane);
      v16bf a1 = load_frag_a(&As[wm * 32 + 16][kk], LDA, lane);
#pragma unroll
      for (int nt = 0; nt < 4; ++nt) {
        v16bf b = load_frag_b(&Bs[wn * 64 + nt * 16][kk], LDA, lane);
        acc[0][nt] = wmma_bf16(a0, b, acc[0][nt]);
        acc[1][nt] = wmma_bf16(a1, b, acc[1][nt]);
      }
    }
    __syncthreads();
  }

  // epilogue: C/D layout -> row = reg + (lane>=16 ? 8:0), col = lane&15
  const int rbase = (lane >> 4) << 3;
  const int colin = lane & 15;
#pragma unroll
  for (int mt = 0; mt < 2; ++mt) {
#pragma unroll
    for (int nt = 0; nt < 4; ++nt) {
#pragma unroll
      for (int r = 0; r < 8; ++r) {
        int m = bm + wm * 32 + mt * 16 + rbase + r;
        int n = bn + wn * 64 + nt * 16 + colin;
        float val = acc[mt][nt][r];
        if (MODE == 0) {
          int h = n / 192;
          int rem = n - h * 192;
          int sel = rem >> 6;
          int d = rem & 63;
          int b_ = m >> 11;          // / TSEQ
          int t  = m & (TSEQ - 1);
          size_t o = (((size_t)(b_ * N_HEAD + h) * TSEQ) + t) * HEAD + d;
          if (sel == 0)      q_ws[o] = (bf16)(val * 0.125f);  // 1/sqrt(64)
          else if (sel == 1) k_ws[o] = (bf16)val;
          else               v_ws[o] = (bf16)val;
        } else {
          out[(size_t)m * N + n] = val + bias[n];
        }
      }
    }
  }
}

// ---------------------------------------------------------------------------
// Flash attention (causal). One block = one (b,h) x 128-query tile, 8 waves.
// q/k/v are bf16 (B,H,T,HEAD); q pre-scaled by 1/sqrt(HEAD).
// ---------------------------------------------------------------------------
__global__ __launch_bounds__(256) void flash_attn_kernel(
    const bf16* __restrict__ q, const bf16* __restrict__ k,
    const bf16* __restrict__ v, bf16* __restrict__ attn_out) {
  constexpr int BQ = 128, BK = 64;
  constexpr int QS = HEAD + 8;   // padded LDS row stride (elements)
  constexpr int PS = BK + 8;
  __shared__ bf16 Qs[BQ][QS];
  __shared__ bf16 Ks[BK][QS];
  __shared__ bf16 Vt[HEAD][PS];  // transposed: [d][key]
  __shared__ bf16 Ps[BQ][PS];    // probability staging (per-wave rows)

  const int tid  = threadIdx.x;
  const int lane = tid & 31;
  const int wv   = tid >> 5;
  const int bh   = blockIdx.x;           // b*N_HEAD + h
  const int qt   = blockIdx.y;
  const int qbase = qt * BQ;

  const bf16* qp = q + (size_t)bh * TSEQ * HEAD;
  const bf16* kp = k + (size_t)bh * TSEQ * HEAD;
  const bf16* vp = v + (size_t)bh * TSEQ * HEAD;
  const unsigned int qs_base = lds_addr_of(&Qs[0][0]);
  const unsigned int ks_base = lds_addr_of(&Ks[0][0]);

  // async-stage Q tile: 128 rows x 8 x 16B chunks
#pragma unroll
  for (int i = 0; i < 4; ++i) {
    int idx = tid + i * 256;
    int r = idx >> 3, ch = idx & 7;
    async_copy16(qs_base + (unsigned)(r * QS * 2 + ch * 16),
                 (unsigned long long)(const char*)qp +
                     ((size_t)(qbase + r) * HEAD + ch * 8) * sizeof(bf16));
  }

  const int rbase = (lane >> 4) << 3;
  const int colin = lane & 15;
  const int qrow  = wv * 16;        // this wave's 16 query rows in the tile
  const int qlo   = qbase + qrow;   // first (smallest) query row of this wave
  const int qhi   = qlo + 15;       // last query row of this wave

  float m_i[8], l_i[8];
  v8f o[4];
#pragma unroll
  for (int r = 0; r < 8; ++r) { m_i[r] = -1e30f; l_i[r] = 0.f; }
#pragma unroll
  for (int nd = 0; nd < 4; ++nd) o[nd] = zero8();

  const int jmax = 2 * qt + 1;  // causal: keys up to qbase+127
  for (int j = 0; j <= jmax; ++j) {
    const int kb = j * BK;
    __syncthreads();  // everyone done reading previous K/V tiles
    // async-stage K tile (64x64 row-major): rows = B-fragment columns of K^T
#pragma unroll
    for (int i = 0; i < 2; ++i) {
      int idx = tid + i * 256;
      int r = idx >> 3, ch = idx & 7;
      async_copy16(ks_base + (unsigned)(r * QS * 2 + ch * 16),
                   (unsigned long long)(const char*)kp +
                       ((size_t)(kb + r) * HEAD + ch * 8) * sizeof(bf16));
    }
    // stage V transposed: Vt[d][key] (manual; async copy cannot transpose)
#pragma unroll
    for (int i = 0; i < 16; ++i) {
      int idx = tid + i * 256;
      int r = idx >> 6, c = idx & 63;
      Vt[c][r] = vp[(size_t)(kb + r) * HEAD + c];
    }
    wait_async();
    __syncthreads();

    // Per-wave causal classification for this key tile (wave-uniform):
    //  kb > qhi            -> tile entirely above diagonal: nothing to do
    //  kb + BK - 1 <= qlo  -> tile entirely valid: skip masking
    if (kb > qhi) continue;  // barriers already done for this iteration
    const bool need_mask = (kb + BK - 1) > qlo;

    // S = Q * K^T  (16 rows x 64 keys per wave)
    v8f s[4];
#pragma unroll
    for (int nt = 0; nt < 4; ++nt) s[nt] = zero8();
    v16bf aq0 = load_frag_a(&Qs[qrow][0], QS, lane);
    v16bf aq1 = load_frag_a(&Qs[qrow][32], QS, lane);
#pragma unroll
    for (int nt = 0; nt < 4; ++nt) {
      v16bf b0 = load_frag_b(&Ks[nt * 16][0], QS, lane);
      v16bf b1 = load_frag_b(&Ks[nt * 16][32], QS, lane);
      s[nt] = wmma_bf16(aq0, b0, s[nt]);
      s[nt] = wmma_bf16(aq1, b1, s[nt]);
    }

    // causal mask (only on diagonal-straddling tiles; s_cbranch-guarded)
    if (need_mask) {
#pragma unroll
      for (int nt = 0; nt < 4; ++nt) {
#pragma unroll
        for (int r = 0; r < 8; ++r) {
          int qg = qlo + rbase + r;
          int kg = kb + nt * 16 + colin;
          if (kg > qg) s[nt][r] = -1e30f;
        }
      }
    }

    // online softmax per query row; write P tile to LDS as bf16
#pragma unroll
    for (int r = 0; r < 8; ++r) {
      float mx = fmaxf(fmaxf(s[0][r], s[1][r]), fmaxf(s[2][r], s[3][r]));
#pragma unroll
      for (int off = 1; off < 16; off <<= 1)
        mx = fmaxf(mx, __shfl_xor(mx, off, 32));
      float mnew = fmaxf(m_i[r], mx);
      float corr = __expf(m_i[r] - mnew);
      float rs = 0.f;
#pragma unroll
      for (int nt = 0; nt < 4; ++nt) {
        float p = __expf(s[nt][r] - mnew);
        s[nt][r] = p;
        rs += p;
      }
#pragma unroll
      for (int off = 1; off < 16; off <<= 1)
        rs += __shfl_xor(rs, off, 32);
      l_i[r] = l_i[r] * corr + rs;
      m_i[r] = mnew;
#pragma unroll
      for (int nd = 0; nd < 4; ++nd) o[nd][r] *= corr;
#pragma unroll
      for (int nt = 0; nt < 4; ++nt)
        Ps[qrow + rbase + r][nt * 16 + colin] = (bf16)s[nt][r];
    }

    // O += P * V
    v16bf p0 = load_frag_a(&Ps[qrow][0], PS, lane);
    v16bf p1 = load_frag_a(&Ps[qrow][32], PS, lane);
#pragma unroll
    for (int nd = 0; nd < 4; ++nd) {
      v16bf b0 = load_frag_b(&Vt[nd * 16][0], PS, lane);
      v16bf b1 = load_frag_b(&Vt[nd * 16][32], PS, lane);
      o[nd] = wmma_bf16(p0, b0, o[nd]);
      o[nd] = wmma_bf16(p1, b1, o[nd]);
    }
  }

  // normalize + write (B,T,C) bf16 for the output projection
  const int b_ = bh >> 4;        // / N_HEAD
  const int h_ = bh & 15;
#pragma unroll
  for (int r = 0; r < 8; ++r) {
    float inv = 1.f / l_i[r];
    int t = qbase + qrow + rbase + r;
#pragma unroll
    for (int nd = 0; nd < 4; ++nd) {
      int c = h_ * HEAD + nd * 16 + colin;
      attn_out[((size_t)b_ * TSEQ + t) * D_MODEL + c] = (bf16)(o[nd][r] * inv);
    }
  }
}

// ---------------------------------------------------------------------------
extern "C" void kernel_launch(void* const* d_in, const int* in_sizes, int n_in,
                              void* d_out, int out_size, void* d_ws,
                              size_t ws_size, hipStream_t stream) {
  const float* x      = (const float*)d_in[0];  // (2,2048,1024)
  const float* qkv_w  = (const float*)d_in[1];  // (3072,1024)
  const float* out_w  = (const float*)d_in[2];  // (1024,1024)
  const float* out_b  = (const float*)d_in[3];  // (1024,)
  float* out          = (float*)d_out;          // (2,2048,1024)

  constexpr size_t PER_HEAD_ELEMS =
      (size_t)BATCH * N_HEAD * TSEQ * HEAD;     // 4,194,304 each
  bf16* q_ws    = (bf16*)d_ws;
  bf16* k_ws    = q_ws + PER_HEAD_ELEMS;
  bf16* v_ws    = k_ws + PER_HEAD_ELEMS;
  bf16* attn_ws = v_ws + PER_HEAD_ELEMS;        // (4096,1024) bf16

  constexpr int M = BATCH * TSEQ;               // 4096

  // 1) fused QKV projection -> per-head q/k/v (q scaled by 1/8)
  dim3 g1(M / 128, (3 * D_MODEL) / 128);        // 32 x 24
  gemm_wmma_kernel<float, 0><<<g1, 256, 0, stream>>>(
      x, qkv_w, M, 3 * D_MODEL, D_MODEL, q_ws, k_ws, v_ws, nullptr, nullptr);

  // 2) causal flash attention
  dim3 g2(BATCH * N_HEAD, TSEQ / 128);          // 32 x 16
  flash_attn_kernel<<<g2, 256, 0, stream>>>(q_ws, k_ws, v_ws, attn_ws);

  // 3) output projection + bias
  dim3 g3(M / 128, D_MODEL / 128);              // 32 x 8
  gemm_wmma_kernel<bf16, 1><<<g3, 256, 0, stream>>>(
      attn_ws, out_w, M, D_MODEL, D_MODEL, nullptr, nullptr, nullptr, out_b,
      out);
}
